// BLSTM_NMC_14869176779116
// MI455X (gfx1250) — matrix operand order repaired and checked by reading, not compiled, not run on hardware
//
#include <hip/hip_runtime.h>
#include <math.h>

typedef __attribute__((ext_vector_type(16))) __bf16 v16bf;
typedef __attribute__((ext_vector_type(8)))  float  v8f;

#define Bsz 512
#define Ssz 128
#define Hsz 512
#define G4  2048  // 4*H

__device__ __forceinline__ unsigned short f32_to_bf16(float f) {
    union { float f; unsigned int u; } cv; cv.f = f;
    unsigned int u = cv.u;
    u += 0x7FFFu + ((u >> 16) & 1u);   // round-to-nearest-even
    return (unsigned short)(u >> 16);
}
__device__ __forceinline__ float softplusf(float x) { return log1pf(expf(x)); }
__device__ __forceinline__ float sigmoidf(float x)  { return 1.0f / (1.0f + expf(-x)); }

// Sample W_hh = mu + softplus(rho)*eps, store TRANSPOSED bf16: Wt[n*512 + k]
__global__ __launch_bounds__(256)
void prep_whh_kernel(const float* __restrict__ mu, const float* __restrict__ rho,
                     const float* __restrict__ eps, unsigned short* __restrict__ Wt) {
    int idx = blockIdx.x * 256 + threadIdx.x;          // idx = n*512 + k
    if (idx >= G4 * Hsz) return;
    int n = idx >> 9, k = idx & (Hsz - 1);
    int src = k * G4 + n;                              // W_hh is (H, 4H) row-major
    float w = mu[src] + softplusf(rho[src]) * eps[src];
    Wt[idx] = f32_to_bf16(w);
}

// Sample W_ih (IN=1 -> 2048 vec) and bias (2048 vec)
__global__ __launch_bounds__(256)
void prep_vec_kernel(const float* __restrict__ wih_mu, const float* __restrict__ wih_rho,
                     const float* __restrict__ wih_eps,
                     const float* __restrict__ b_mu, const float* __restrict__ b_rho,
                     const float* __restrict__ b_eps,
                     float* __restrict__ wih_s, float* __restrict__ bias_s) {
    int i = blockIdx.x * 256 + threadIdx.x;
    if (i >= G4) return;
    wih_s[i]  = wih_mu[i] + softplusf(wih_rho[i]) * wih_eps[i];
    bias_s[i] = b_mu[i]   + softplusf(b_rho[i])   * b_eps[i];
}

// xd[b,s] = x[b,s] * mask_in[b,s]
__global__ __launch_bounds__(256)
void prep_x_kernel(const float* __restrict__ x, const float* __restrict__ mask_in,
                   float* __restrict__ xd) {
    int i = blockIdx.x * 256 + threadIdx.x;
    if (i < Bsz * Ssz) xd[i] = x[i] * mask_in[i];
}

// zero c-state and both bf16 h ping-pong buffers
__global__ __launch_bounds__(256)
void init_state_kernel(float* __restrict__ c, unsigned short* __restrict__ hb0,
                       unsigned short* __restrict__ hb1) {
    int i = blockIdx.x * 256 + threadIdx.x;
    if (i < Bsz * Hsz) { c[i] = 0.0f; hb0[i] = 0; hb1[i] = 0; }
}

// One LSTM timestep: gates = h_in @ W_hh + x[t]*W_ih + b, fused cell update.
// grid (8,8): blockIdx.x = 64-col tile (within H), blockIdx.y = 64-row tile (batch).
// 8 waves: wave = 2(m) x 4(n); each wave: 2 m-subtiles x 4 gates of 16x16 WMMA tiles.
__global__ __launch_bounds__(256)
void lstm_step_kernel(const unsigned short* __restrict__ h_in,
                      unsigned short* __restrict__ h_out,
                      float* __restrict__ c_state,
                      float* __restrict__ h_f32,
                      const unsigned short* __restrict__ Wt,
                      const float* __restrict__ wih,
                      const float* __restrict__ bias,
                      const float* __restrict__ xd,
                      int t) {
    const int lane = threadIdx.x & 31;
    const int wave = threadIdx.x >> 5;
    const int mw = wave & 1;          // 0..1
    const int nw = wave >> 1;         // 0..3
    const int m0 = blockIdx.y * 64 + mw * 32;   // covers rows m0..m0+31 (two 16-row tiles)
    const int n0 = blockIdx.x * 64 + nw * 16;   // 16 columns per wave (per gate)
    const int lrow = lane & 15;
    const int half = lane >> 4;

    v8f acc[2][4];
    #pragma unroll
    for (int mi = 0; mi < 2; ++mi)
        #pragma unroll
        for (int g = 0; g < 4; ++g)
            acc[mi][g] = (v8f){0.f,0.f,0.f,0.f,0.f,0.f,0.f,0.f};

    union Frag { v16bf v; int4 q[2]; };
    const int kbA = half * 16;   // A: lanes16-31 hold K+16.. (same k order as B)
    const int kbB = half * 16;   // B: lanes16-31 hold K+16.. (same k order as A)

    for (int k0 = 0; k0 < Hsz; k0 += 32) {
        Frag a[2];
        #pragma unroll
        for (int mi = 0; mi < 2; ++mi) {
            const unsigned short* p = h_in + (size_t)(m0 + mi * 16 + lrow) * Hsz + k0 + kbA;
            a[mi].q[0] = *reinterpret_cast<const int4*>(p);        // K = kbA..kbA+7
            a[mi].q[1] = *reinterpret_cast<const int4*>(p + 8);    // K = kbA+8..kbA+15
        }
        Frag b[4];
        #pragma unroll
        for (int g = 0; g < 4; ++g) {
            const unsigned short* p = Wt + (size_t)(n0 + lrow + g * Hsz) * Hsz + k0 + kbB;
            b[g].q[0] = *reinterpret_cast<const int4*>(p);         // K = kbB..kbB+7
            b[g].q[1] = *reinterpret_cast<const int4*>(p + 8);     // K = kbB+8..kbB+15
        }
        #pragma unroll
        for (int mi = 0; mi < 2; ++mi)
            #pragma unroll
            for (int g = 0; g < 4; ++g)
                acc[mi][g] = __builtin_amdgcn_wmma_f32_16x16x32_bf16(
                    false, a[mi].v, false, b[g].v, (short)0, acc[mi][g], false, false);
    }

    // fused pointwise LSTM cell update
    const int nn = n0 + lrow;
    float wv[4], bv[4];
    #pragma unroll
    for (int g = 0; g < 4; ++g) { wv[g] = wih[nn + g * Hsz]; bv[g] = bias[nn + g * Hsz]; }

    #pragma unroll
    for (int mi = 0; mi < 2; ++mi) {
        #pragma unroll
        for (int r = 0; r < 8; ++r) {
            const int m = m0 + mi * 16 + half * 8 + r;   // C/D layout: VGPR r -> row r (+8 for upper lanes)
            const float xv = xd[m * Ssz + t];
            float ig = acc[mi][0][r] + xv * wv[0] + bv[0];
            float fg = acc[mi][1][r] + xv * wv[1] + bv[1];
            float gg = acc[mi][2][r] + xv * wv[2] + bv[2];
            float og = acc[mi][3][r] + xv * wv[3] + bv[3];
            float it = sigmoidf(ig), ft = sigmoidf(fg);
            float gt = tanhf(gg),   ot = sigmoidf(og);
            const int idx = m * Hsz + nn;
            float cn = ft * c_state[idx] + it * gt;
            float hn = ot * tanhf(cn);
            c_state[idx] = cn;
            h_f32[idx]   = hn;
            h_out[idx]   = f32_to_bf16(hn);
        }
    }
}

// out[b] = sum_h h_last[b,h]*mask_out[b,h]*W_lin[h] + b_lin
__global__ __launch_bounds__(256)
void head_kernel(const float* __restrict__ h, const float* __restrict__ mask_out,
                 const float* __restrict__ wlin, const float* __restrict__ blin,
                 float* __restrict__ out) {
    __shared__ float sm[256];
    const int b = blockIdx.x, tid = threadIdx.x;
    float s = 0.f;
    for (int j = tid; j < Hsz; j += 256)
        s += h[b * Hsz + j] * mask_out[b * Hsz + j] * wlin[j];
    sm[tid] = s; __syncthreads();
    for (int off = 128; off > 0; off >>= 1) {
        if (tid < off) sm[tid] += sm[tid + off];
        __syncthreads();
    }
    if (tid == 0) out[b] = sm[0] + blin[0];
}

extern "C" void kernel_launch(void* const* d_in, const int* in_sizes, int n_in,
                              void* d_out, int out_size, void* d_ws, size_t ws_size,
                              hipStream_t stream) {
    (void)in_sizes; (void)n_in; (void)out_size; (void)ws_size;
    const float* x        = (const float*)d_in[0];
    const float* Wih_mu   = (const float*)d_in[1];
    const float* Wih_rho  = (const float*)d_in[2];
    const float* eps_ih   = (const float*)d_in[3];
    const float* Whh_mu   = (const float*)d_in[4];
    const float* Whh_rho  = (const float*)d_in[5];
    const float* eps_hh   = (const float*)d_in[6];
    const float* b_mu     = (const float*)d_in[7];
    const float* b_rho    = (const float*)d_in[8];
    const float* eps_b    = (const float*)d_in[9];
    const float* W_lin    = (const float*)d_in[10];
    const float* b_lin    = (const float*)d_in[11];
    const float* mask_in  = (const float*)d_in[12];
    const float* mask_out = (const float*)d_in[13];
    float* out = (float*)d_out;

    // workspace layout
    char* w = (char*)d_ws;
    unsigned short* Wt = (unsigned short*)w;  w += (size_t)G4 * Hsz * sizeof(unsigned short); // 2 MB
    float* wih_s  = (float*)w;                w += (size_t)G4 * sizeof(float);
    float* bias_s = (float*)w;                w += (size_t)G4 * sizeof(float);
    float* xd     = (float*)w;                w += (size_t)Bsz * Ssz * sizeof(float);
    float* hF     = (float*)w;                w += (size_t)Bsz * Hsz * sizeof(float);
    float* cS     = (float*)w;                w += (size_t)Bsz * Hsz * sizeof(float);
    unsigned short* hb0 = (unsigned short*)w; w += (size_t)Bsz * Hsz * sizeof(unsigned short);
    unsigned short* hb1 = (unsigned short*)w;

    prep_whh_kernel<<<(G4 * Hsz + 255) / 256, 256, 0, stream>>>(Whh_mu, Whh_rho, eps_hh, Wt);
    prep_vec_kernel<<<(G4 + 255) / 256, 256, 0, stream>>>(Wih_mu, Wih_rho, eps_ih,
                                                          b_mu, b_rho, eps_b, wih_s, bias_s);
    prep_x_kernel<<<(Bsz * Ssz + 255) / 256, 256, 0, stream>>>(x, mask_in, xd);
    init_state_kernel<<<(Bsz * Hsz + 255) / 256, 256, 0, stream>>>(cS, hb0, hb1);

    dim3 grid(Hsz / 64, Bsz / 64);  // (8, 8)
    for (int t = 0; t < Ssz; ++t) {
        const unsigned short* hin = (t & 1) ? hb1 : hb0;
        unsigned short*       hout = (t & 1) ? hb0 : hb1;
        lstm_step_kernel<<<grid, 256, 0, stream>>>(hin, hout, cS, hF, Wt, wih_s, bias_s, xd, t);
    }

    head_kernel<<<Bsz, 256, 0, stream>>>(hF, mask_out, W_lin, b_lin, out);
}
